// SAGPool_30047591203115
// MI455X (gfx1250) — matrix operand order repaired and checked by reading, not compiled, not run on hardware
//
#include <hip/hip_runtime.h>
#include <hip/hip_bf16.h>

#define NUM_GRAPHS       64
#define NODES_PER_GRAPH  2048
#define NUM_NODES        131072      // 64 * 2048
#define D_FEAT           256
#define NUM_EDGES        4194304     // NUM_NODES * 32
#define KSEL             1024        // kept nodes per graph
#define NUM_KEPT         (NUM_GRAPHS * KSEL)   // 65536

// d_out layout (floats), reference return order:
// x_new[65536*256] | edge_index_new[2*E] | edge_attr_new[E] | batch_new[65536]
// | perm[65536] | score[131072] | edge_mask[E]
#define OFF_XNEW   ((size_t)0)
#define OFF_EI     ((size_t)16777216)
#define OFF_EA     (OFF_EI + (size_t)2 * NUM_EDGES)
#define OFF_BATCH  (OFF_EA + (size_t)NUM_EDGES)
#define OFF_PERM   (OFF_BATCH + (size_t)NUM_KEPT)
#define OFF_SCORE  (OFF_PERM + (size_t)NUM_KEPT)
#define OFF_MASK   (OFF_SCORE + (size_t)NUM_NODES)

typedef __attribute__((ext_vector_type(16))) _Float16 v16h;
typedef __attribute__((ext_vector_type(8)))  float    v8f;

// ---------------------------------------------------------------------------
// Kernel 1: score = sigmoid(sigmoid(x @ W + b)) via V_WMMA_F32_16X16X32_F16.
// One wave32 computes one 16-node tile; B holds W replicated across all 16
// columns, so every column of D equals the 16 node scores. K-loop: 8 x 32.
// A-matrix 16x32 f16 layout (ISA 7.12.2): lanes 0-15 => M=0..15,
//   VGPR d in 0..3: K = 2d,2d+1 (+8 for lanes 16-31); d in 4..7: K += 16.
// ---------------------------------------------------------------------------
__global__ void sag_score_wmma(const float* __restrict__ x,
                               const float* __restrict__ W,
                               const float* __restrict__ bias_p,
                               float* __restrict__ score)
{
    const int wave = (blockIdx.x * blockDim.x + threadIdx.x) >> 5;
    const int lane = threadIdx.x & 31;
    const int m    = lane & 15;
    const int hi   = lane >> 4;                 // 0: K-low half, 1: K-high half
    const int tile = wave;                      // 16 nodes per tile
    const float* __restrict__ xrow = x + (size_t)(tile * 16 + m) * D_FEAT;

    v8f c = {};
#pragma unroll
    for (int kk = 0; kk < 8; ++kk) {
        const int kbase = kk * 32;
        v16h a, b;
#pragma unroll
        for (int d = 0; d < 8; ++d) {
            const int koffA = ((d < 4) ? (2 * d) : (16 + 2 * (d - 4))) + 8 * hi;
            a[2 * d]     = (_Float16)xrow[kbase + koffA];
            a[2 * d + 1] = (_Float16)xrow[kbase + koffA + 1];
            const int koffB = ((d < 4) ? (2 * d) : (16 + 2 * (d - 4))) + 8 * hi;  // column-replicated W
            b[2 * d]     = (_Float16)W[kbase + koffB];
            b[2 * d + 1] = (_Float16)W[kbase + koffB + 1];
        }
        c = __builtin_amdgcn_wmma_f32_16x16x32_f16(
                /*neg_a=*/false, a, /*neg_b=*/false, b,
                /*c_mod=*/(short)0, c, /*reuse_a=*/false, /*reuse_b=*/false);
    }

    const float bias = bias_p[0];
    // D layout: VGPR r, lanes 0-15 -> M=r, lanes 16-31 -> M=r+8 (all N equal).
    if ((lane & 15) == 0) {
#pragma unroll
        for (int r = 0; r < 8; ++r) {
            float s = c[r] + bias;
            s = 1.0f / (1.0f + __expf(-s));   // non_linearity
            s = 1.0f / (1.0f + __expf(-s));   // sigmoid(score - 0)
            score[tile * 16 + hi * 8 + r] = s;
        }
    }
}

// ---------------------------------------------------------------------------
// Kernel 2: per-graph top-1024 via full LDS bitonic sort of 2048 pairs,
// descending by score, ascending index tie-break (matches lax.top_k order).
// ---------------------------------------------------------------------------
__global__ void sag_topk(const float* __restrict__ score,
                         float* __restrict__ perm_f,
                         float* __restrict__ batch_new_f,
                         int* __restrict__ perm_i)
{
    __shared__ float s[NODES_PER_GRAPH];
    __shared__ int   id[NODES_PER_GRAPH];
    const int g   = blockIdx.x;
    const int tid = threadIdx.x;

    for (int t = tid; t < NODES_PER_GRAPH; t += 1024) {
        s[t]  = score[g * NODES_PER_GRAPH + t];
        id[t] = t;
    }
    __syncthreads();

    for (int k = 2; k <= NODES_PER_GRAPH; k <<= 1) {
        for (int j = k >> 1; j > 0; j >>= 1) {
            for (int t = tid; t < NODES_PER_GRAPH; t += 1024) {
                const int ixj = t ^ j;
                if (ixj > t) {
                    const bool desc = ((t & k) == 0);
                    const float si = s[t],  sj = s[ixj];
                    const int   ii = id[t], ij = id[ixj];
                    // "wrong order" for a descending segment:
                    const bool bad = (si < sj) || (si == sj && ii > ij);
                    if (desc ? bad : !bad) {
                        s[t] = sj; s[ixj] = si;
                        id[t] = ij; id[ixj] = ii;
                    }
                }
            }
            __syncthreads();
        }
    }

    if (tid < KSEL) {
        const int gidx = g * NODES_PER_GRAPH + id[tid];
        perm_i[g * KSEL + tid]      = gidx;
        perm_f[g * KSEL + tid]      = (float)gidx;
        batch_new_f[g * KSEL + tid] = (float)g;
    }
}

// ---------------------------------------------------------------------------
// Kernels 3a/3b: inverse mapping (node -> new id, -1 if dropped).
// ---------------------------------------------------------------------------
__global__ void sag_map_init(int* __restrict__ mapping)
{
    const int i = blockIdx.x * blockDim.x + threadIdx.x;
    if (i < NUM_NODES) mapping[i] = -1;
}

__global__ void sag_map_set(const int* __restrict__ perm_i,
                            int* __restrict__ mapping)
{
    const int i = blockIdx.x * blockDim.x + threadIdx.x;
    if (i < NUM_KEPT) mapping[perm_i[i]] = i;
}

// ---------------------------------------------------------------------------
// Kernel 4: filter_adj — relabel edges, mask dropped ones.
// ---------------------------------------------------------------------------
__global__ void sag_edge_filter(const int* __restrict__ ei,
                                const float* __restrict__ ea,
                                const int* __restrict__ mapping,
                                float* __restrict__ ei_out,
                                float* __restrict__ ea_out,
                                float* __restrict__ mask_out)
{
    const int e = blockIdx.x * blockDim.x + threadIdx.x;
    if (e >= NUM_EDGES) return;
    const int src = ei[e];
    const int dst = ei[NUM_EDGES + e];
    const int r = mapping[src];
    const int c = mapping[dst];
    const bool keep = (r >= 0) && (c >= 0);
    ei_out[e]             = keep ? (float)r : 0.0f;
    ei_out[NUM_EDGES + e] = keep ? (float)c : 0.0f;
    ea_out[e]             = keep ? ea[e] : 0.0f;
    mask_out[e]           = keep ? 1.0f : 0.0f;
}

// ---------------------------------------------------------------------------
// Kernel 5: x_new = x[perm], float4 row copies (global_load/store_b128).
// ---------------------------------------------------------------------------
__global__ void sag_gather_x(const float* __restrict__ x,
                             const int* __restrict__ perm_i,
                             float* __restrict__ xout)
{
    const int row  = blockIdx.x;        // 65536 rows
    const int t    = threadIdx.x;       // 64 threads * float4 = 256 floats
    const int node = perm_i[row];
    const float4* __restrict__ src = (const float4*)(x + (size_t)node * D_FEAT);
    float4* __restrict__ dst       = (float4*)(xout + (size_t)row * D_FEAT);
    dst[t] = src[t];
}

// ---------------------------------------------------------------------------
extern "C" void kernel_launch(void* const* d_in, const int* in_sizes, int n_in,
                              void* d_out, int out_size, void* d_ws, size_t ws_size,
                              hipStream_t stream)
{
    const float* x    = (const float*)d_in[0];
    const int*   ei   = (const int*)  d_in[1];
    const float* ea   = (const float*)d_in[2];
    // d_in[3] = batch (unused: graphs are contiguous, batch_new derived from g)
    const float* W    = (const float*)d_in[4];
    const float* bias = (const float*)d_in[5];

    float* out = (float*)d_out;
    float* out_xnew  = out + OFF_XNEW;
    float* out_ei    = out + OFF_EI;
    float* out_ea    = out + OFF_EA;
    float* out_batch = out + OFF_BATCH;
    float* out_perm  = out + OFF_PERM;
    float* out_score = out + OFF_SCORE;
    float* out_mask  = out + OFF_MASK;

    int* perm_i  = (int*)d_ws;               // 65536 ints
    int* mapping = perm_i + NUM_KEPT;        // 131072 ints

    // 1) scores: 8192 node-tiles of 16, 8 waves (256 thr) per block
    sag_score_wmma<<<(NUM_NODES / 16) / 8, 256, 0, stream>>>(x, W, bias, out_score);

    // 2) per-graph top-k (bitonic sort in LDS)
    sag_topk<<<NUM_GRAPHS, 1024, 0, stream>>>(out_score, out_perm, out_batch, perm_i);

    // 3) inverse mapping
    sag_map_init<<<(NUM_NODES + 255) / 256, 256, 0, stream>>>(mapping);
    sag_map_set<<<(NUM_KEPT + 255) / 256, 256, 0, stream>>>(perm_i, mapping);

    // 4) edge filtering / relabeling
    sag_edge_filter<<<(NUM_EDGES + 255) / 256, 256, 0, stream>>>(
        ei, ea, mapping, out_ei, out_ea, out_mask);

    // 5) x gather
    sag_gather_x<<<NUM_KEPT, 64, 0, stream>>>(x, perm_i, out_xnew);
}
